// SparseAttention_39298950758496
// MI455X (gfx1250) — matrix operand order repaired and checked by reading, not compiled, not run on hardware
//
#include <hip/hip_runtime.h>

#define SEQ     8192
#define HID     1024
#define NHEADS  16
#define HD      64
#define WIN     128
#define NCHUNK  (SEQ / WIN)   // 64
#define KW      384           // keys in the 3w window per chunk
#define VSTR    392           // padded LDS stride (ushorts) for V^T rows

typedef __attribute__((ext_vector_type(16))) __bf16 v16bf;
typedef __attribute__((ext_vector_type(8)))  __bf16 v8bf;
typedef __attribute__((ext_vector_type(4)))  __bf16 v4bf;
typedef __attribute__((ext_vector_type(16))) float  v16f;
typedef __attribute__((ext_vector_type(8)))  float  v8f;
typedef __attribute__((ext_vector_type(4)))  float  v4f;

__device__ __forceinline__ float redmax16(float x) {
#pragma unroll
  for (int off = 1; off < 16; off <<= 1)
    x = fmaxf(x, __shfl_xor(x, off, 32)); // stays within 16-lane half
  return x;
}
__device__ __forceinline__ float redsum16(float x) {
#pragma unroll
  for (int off = 1; off < 16; off <<= 1)
    x += __shfl_xor(x, off, 32);
  return x;
}

__device__ __forceinline__ v8f vzero8() {
  v8f z;
#pragma unroll
  for (int i = 0; i < 8; ++i) z[i] = 0.0f;
  return z;
}

// A-fragment K-index map for 16-bit 16x32 A (contiguous order):
// e=0..15 -> K = e + 16*half
__device__ __forceinline__ int kmapA(int e, int half) {
  return e + half * 16;
}

__device__ __forceinline__ v16bf cat8(v8bf a, v8bf b) {
  return __builtin_shufflevector(a, b, 0, 1, 2, 3, 4, 5, 6, 7,
                                        8, 9, 10, 11, 12, 13, 14, 15);
}

template <bool EDGE>
__device__ __forceinline__ void swa_body(const float* __restrict__ q,
                                         const float* __restrict__ k,
                                         const float* __restrict__ v,
                                         float* __restrict__ out,
                                         unsigned short* Vt,
                                         unsigned short* Plds,
                                         int c, int h, int b) {
  const int tid  = threadIdx.x;
  const int wv   = tid >> 5;
  const int lane = tid & 31;
  const int n    = lane & 15;
  const int half = lane >> 4;

  const long long bsBase = (long long)b * SEQ;
  const int key0 = c * WIN - WIN;

  // ---- Stage V chunk transposed into LDS as bf16: Vt[d][key], padded ----
  for (int idx = tid; idx < KW * (HD / 4); idx += 256) {
    int row = idx >> 4;          // key index in window
    int c4  = idx & 15;          // group of 4 head-dims
    int kp  = key0 + row;
    v4f f;
#pragma unroll
    for (int j = 0; j < 4; ++j) f[j] = 0.0f;
    if (!EDGE || ((unsigned)kp < SEQ))
      f = *(const v4f*)(v + (bsBase + kp) * HID + h * HD + c4 * 4);
    v4bf bf = __builtin_convertvector(f, v4bf);
    union { v4bf v_; unsigned short u[4]; } cv; cv.v_ = bf;
#pragma unroll
    for (int j = 0; j < 4; ++j)
      Vt[(c4 * 4 + j) * VSTR + row] = cv.u[j];
  }
  __syncthreads();

  // ---- Q A-fragments (two K=32 steps over head dim) ----
  const int qrow = c * WIN + wv * 16 + n;   // A layout: row = lane % 16
  const float* qp = q + (bsBase + qrow) * HID + h * HD;
  v16f qf0, qf1;
#pragma unroll
  for (int e = 0; e < 16; ++e) {
    int dk = kmapA(e, half);
    qf0[e] = qp[dk];
    qf1[e] = qp[32 + dk];
  }
  v16bf a0 = __builtin_convertvector(qf0, v16bf);
  v16bf a1 = __builtin_convertvector(qf1, v16bf);

  float m[8], l[8];
  v8f Oc[4];
#pragma unroll
  for (int i = 0; i < 8; ++i) { m[i] = -1e30f; l[i] = 0.0f; }
#pragma unroll
  for (int nt = 0; nt < 4; ++nt) Oc[nt] = vzero8();

  const int wvbase = wv * 512;                       // P scratch (ushorts)
  const int qbase  = c * WIN + wv * 16 + half * 8;   // D-layout row base
  const int dbase  = key0 + n - qbase;               // diff = dbase + tt*16 - i

  // Band is +-128 around this wave's rows -> only tiles wv..wv+16 matter.
  const int tplo = wv >> 1;
  const int tphi = (wv + 16) >> 1;

  for (int tp = tplo; tp <= tphi; ++tp) {
    if (tp < tphi) {                      // prefetch next pair's K rows
      int kpn = key0 + (tp + 1) * 32 + n;
      if ((unsigned)kpn < SEQ)
        __builtin_prefetch(k + (bsBase + kpn) * HID + h * HD, 0, 1);
    }

    float s[2][8];
#pragma unroll
    for (int ht = 0; ht < 2; ++ht) {
      const int tt = tp * 2 + ht;
      const int kp = key0 + tt * 16 + n;       // this lane's key position
      const bool kin = !EDGE || ((unsigned)kp < SEQ);
      v16bf b0, b1;
      if (kin) {
        // B layout (32x16): element e -> K = half*16 + e (per K-step)
        const float* kpp = k + (bsBase + kp) * HID + h * HD + half * 16;
        b0 = __builtin_convertvector(*(const v16f*)kpp, v16bf);
        b1 = __builtin_convertvector(*(const v16f*)(kpp + 32), v16bf);
      } else {
        v16f zf;
#pragma unroll
        for (int e = 0; e < 16; ++e) zf[e] = 0.0f;
        b0 = __builtin_convertvector(zf, v16bf);
        b1 = b0;
      }
      v8f acc = vzero8();
      acc = __builtin_amdgcn_wmma_f32_16x16x32_bf16(false, a0, false, b0,
                                                    (short)0, acc, false, false);
      acc = __builtin_amdgcn_wmma_f32_16x16x32_bf16(false, a1, false, b1,
                                                    (short)0, acc, false, false);
      const int dtt = dbase + tt * 16;
#pragma unroll
      for (int i = 0; i < 8; ++i) {
        bool valid = ((unsigned)(dtt - i + WIN) <= 2u * WIN) && kin;
        s[ht][i] = valid ? acc[i] : -1e30f;
      }
    }

    // ---- online softmax update over these 32 key columns ----
    float scale[8];
    v8f p0v, p1v;
#pragma unroll
    for (int i = 0; i < 8; ++i) {
      float mx = redmax16(fmaxf(s[0][i], s[1][i]));
      float mn = fmaxf(m[i], mx);
      scale[i] = __expf(m[i] - mn);
      m[i] = mn;
      p0v[i] = __expf(s[0][i] - mn);
      p1v[i] = __expf(s[1][i] - mn);
      l[i] = l[i] * scale[i] + redsum16(p0v[i] + p1v[i]);
    }
#pragma unroll
    for (int nt = 0; nt < 4; ++nt)
#pragma unroll
      for (int i = 0; i < 8; ++i) Oc[nt][i] *= scale[i];

    // ---- D-layout -> A-layout transpose of P through per-wave LDS ----
    union { v8bf v_; unsigned short u[8]; } pb0, pb1;
    pb0.v_ = __builtin_convertvector(p0v, v8bf);
    pb1.v_ = __builtin_convertvector(p1v, v8bf);
#pragma unroll
    for (int i = 0; i < 8; ++i) {
      int row = i + half * 8;
      Plds[wvbase + row * 32 + n]      = pb0.u[i];
      Plds[wvbase + row * 32 + 16 + n] = pb1.u[i];
    }
    // Same-wave DS ops are in-order (ISA 7.1); A-map K indices are two runs
    // of 8 consecutive values -> two ds_load_b128 reads.
    const unsigned short* pr = Plds + wvbase + n * 32 + 8 * half;
    v16bf pa = cat8(*(const v8bf*)pr, *(const v8bf*)(pr + 16));

    // ---- P (16x32) x V (32x64): B-fragments in table order (two runs of 8) from V^T ----
#pragma unroll
    for (int nt = 0; nt < 4; ++nt) {
      const unsigned short* vp = Vt + (nt * 16 + n) * VSTR + tp * 32 + half * 8;
      v16bf vb = cat8(*(const v8bf*)vp, *(const v8bf*)(vp + 16));
      Oc[nt] = __builtin_amdgcn_wmma_f32_16x16x32_bf16(false, pa, false, vb,
                                                       (short)0, Oc[nt], false, false);
    }
  }

  // ---- epilogue: normalize and store fp32 ----
#pragma unroll
  for (int i = 0; i < 8; ++i) l[i] = 1.0f / l[i];
#pragma unroll
  for (int nt = 0; nt < 4; ++nt)
#pragma unroll
    for (int i = 0; i < 8; ++i) {
      int row = qbase + i;
      out[(bsBase + row) * HID + h * HD + nt * 16 + n] = Oc[nt][i] * l[i];
    }
}

__global__ __launch_bounds__(256, 1)
void swa_wmma_kernel(const float* __restrict__ q,
                     const float* __restrict__ k,
                     const float* __restrict__ v,
                     float* __restrict__ out) {
  __shared__ __align__(16) unsigned short Vt[HD * VSTR];     // ~49 KB
  __shared__ __align__(16) unsigned short Plds[8 * 512];     //   8 KB

  const int c = blockIdx.x;
  const int h = blockIdx.y;
  const int b = blockIdx.z;

  if (c == 0 || c == NCHUNK - 1)
    swa_body<true>(q, k, v, out, Vt, Plds, c, h, b);
  else
    swa_body<false>(q, k, v, out, Vt, Plds, c, h, b);
}

extern "C" void kernel_launch(void* const* d_in, const int* in_sizes, int n_in,
                              void* d_out, int out_size, void* d_ws, size_t ws_size,
                              hipStream_t stream) {
  const float* q = (const float*)d_in[0];
  const float* k = (const float*)d_in[1];
  const float* v = (const float*)d_in[2];
  float* out = (float*)d_out;
  (void)d_ws; (void)ws_size; (void)n_in; (void)out_size;

  int Bn = in_sizes[0] / (SEQ * HID);   // batch from flat size (=2)
  dim3 grid(NCHUNK, NHEADS, Bn);
  swa_wmma_kernel<<<grid, dim3(256, 1, 1), 0, stream>>>(q, k, v, out);
}
